// TransformerBlock_9904194584652
// MI455X (gfx1250) — matrix operand order repaired and checked by reading, not compiled, not run on hardware
//
#include <hip/hip_runtime.h>
#include <hip/hip_bf16.h>

// ---------------- problem constants ----------------
#define DM    1024
#define NH    16
#define HD    64
#define DFF   4096
#define BATCH 4
#define SEQ   2048
#define MTOT  (BATCH*SEQ)
#define LN_EPS 1e-5f

typedef __bf16 bf16;
typedef __attribute__((ext_vector_type(16))) __bf16 v16bf;
typedef __attribute__((ext_vector_type(8)))  __bf16 v8bf;
typedef __attribute__((ext_vector_type(8)))  float  v8f;
typedef __attribute__((ext_vector_type(4)))  int    v4i;

#define AS1 __attribute__((address_space(1)))
#define AS3 __attribute__((address_space(3)))

// ---- CDNA5 async global->LDS path (guarded; falls back to sync copies) ----
#if defined(__has_builtin)
#  if __has_builtin(__builtin_amdgcn_global_load_async_to_lds_b128)
#    define HAVE_ASYNC_LDS 1
#  endif
#  if __has_builtin(__builtin_amdgcn_tensor_load_to_lds)
#    define HAVE_TDM 1
#  endif
#endif
#ifndef HAVE_ASYNC_LDS
#  define HAVE_ASYNC_LDS 0
#endif
#ifndef HAVE_TDM
#  define HAVE_TDM 0
#endif

#if HAVE_ASYNC_LDS
static __device__ __forceinline__ void wait_async0() {
#if __has_builtin(__builtin_amdgcn_s_wait_asynccnt)
  __builtin_amdgcn_s_wait_asynccnt(0);
#else
  asm volatile("s_wait_asynccnt 0" ::: "memory");
#endif
}
#endif

#if HAVE_TDM
typedef __attribute__((ext_vector_type(4))) unsigned int v4u;
typedef __attribute__((ext_vector_type(8))) int          v8i;

static __device__ __forceinline__ void wait_tensor0() {
#if __has_builtin(__builtin_amdgcn_s_wait_tensorcnt)
  __builtin_amdgcn_s_wait_tensorcnt((short)0);
#else
  asm volatile("s_wait_tensorcnt 0" ::: "memory");
#endif
}

// TDM: DMA one 2-D tile [rows=128 x cols=32 bf16] from a row-major [*, K]
// matrix into LDS (contiguous 128x32). Descriptor per ISA 8.3-8.5:
//   group0: count=1 | lds_addr | global_addr | type=2
//   group1: data_size=2B, tensor_dim0=K, tensor_dim1=128,
//           tile_dim0=32, tile_dim1=128, tensor_dim0_stride=K
//   groups 2/3: zero (2-D tensor)
static __device__ __forceinline__ void tdm_load_tile(const bf16* gbase,
                                                     unsigned lds_off,
                                                     int K_elems) {
  const unsigned long long ga = (unsigned long long)(size_t)gbase;
  v4u g0;
  g0.x = 1u;                                            // count=1 (valid user D#)
  g0.y = lds_off;                                       // lds_addr (bytes)
  g0.z = (unsigned)ga;                                  // global_addr[31:0]
  g0.w = (unsigned)((ga >> 32) & 0x1FFFFFFu) | (2u << 30); // addr[56:32] | type=2
  v8i g1;
  g1[0] = (int)(1u << 16);                              // data_size=1 (2 bytes)
  g1[1] = (int)(((unsigned)K_elems & 0xFFFFu) << 16);   // tensor_dim0[15:0] @bits63:48
  g1[2] = (int)((((unsigned)K_elems >> 16) & 0xFFFFu)   // tensor_dim0[31:16]
                | (128u << 16));                        // tensor_dim1[15:0]
  g1[3] = (int)(32u << 16);                             // tensor_dim1 hi=0 | tile_dim0=32
  g1[4] = 128;                                          // tile_dim1=128, tile_dim2=0
  g1[5] = K_elems;                                      // tensor_dim0_stride[31:0]
  g1[6] = 0;                                            // stride hi | dim1_stride lo
  g1[7] = 0;
  v4i gz = {0, 0, 0, 0};
#if __clang_major__ >= 23
  v8i gz8 = {0, 0, 0, 0, 0, 0, 0, 0};
  __builtin_amdgcn_tensor_load_to_lds(g0, g1, gz, gz, gz8, 0);
#else
  __builtin_amdgcn_tensor_load_to_lds(g0, g1, gz, gz, 0);
#endif
}
#endif  // HAVE_TDM

static __device__ __forceinline__ v8f wmma_bf16(v16bf a, v16bf b, v8f c) {
  // D = A(16x32 bf16) x B(32x16 bf16) + C(16x16 f32)
  return __builtin_amdgcn_wmma_f32_16x16x32_bf16(
      /*neg_a=*/false, a, /*neg_b=*/false, b,
      /*c_mod=*/(short)0, c, /*reuse_a=*/false, /*reuse_b=*/false);
}

// Build an A-fragment style v16bf from two 8-element contiguous chunks.
static __device__ __forceinline__ v16bf join16(const bf16* p0, const bf16* p1) {
  v8bf lo = *(const v8bf*)p0;
  v8bf hi = *(const v8bf*)p1;
  return __builtin_shufflevector(lo, hi, 0,1,2,3,4,5,6,7,8,9,10,11,12,13,14,15);
}

static __device__ __forceinline__ v8f vzero8() {
  v8f z = {0.f,0.f,0.f,0.f,0.f,0.f,0.f,0.f};
  return z;
}

// ---------------- fp32 -> bf16 convert ----------------
__global__ __launch_bounds__(256)
void cvt_f32_bf16_kernel(const float* __restrict__ in, bf16* __restrict__ out, int n) {
  int i = (blockIdx.x * 256 + threadIdx.x) * 4;
  if (i < n) {
    float4 v = *(const float4*)(in + i);
    out[i + 0] = (bf16)v.x;
    out[i + 1] = (bf16)v.y;
    out[i + 2] = (bf16)v.z;
    out[i + 3] = (bf16)v.w;
  }
}

// ---------------- LayerNorm (row of 1024) -> bf16 ----------------
__global__ __launch_bounds__(256)
void layernorm_bf16_kernel(const float* __restrict__ x, const float* __restrict__ g,
                           const float* __restrict__ bta, bf16* __restrict__ out) {
  __shared__ float red[16];
  const int row = blockIdx.x;
  const int tid = threadIdx.x;
  const float4 v = ((const float4*)(x + (long)row * DM))[tid];
  float s = v.x + v.y + v.z + v.w;
  float q = v.x*v.x + v.y*v.y + v.z*v.z + v.w*v.w;
  #pragma unroll
  for (int off = 1; off < 32; off <<= 1) {
    s += __shfl_xor(s, off);
    q += __shfl_xor(q, off);
  }
  if ((tid & 31) == 0) { red[tid >> 5] = s; red[8 + (tid >> 5)] = q; }
  __syncthreads();
  float ts = 0.f, tq = 0.f;
  #pragma unroll
  for (int i = 0; i < 8; ++i) { ts += red[i]; tq += red[8 + i]; }
  const float mu   = ts * (1.0f / DM);
  const float var  = tq * (1.0f / DM) - mu * mu;
  const float rstd = rsqrtf(var + LN_EPS);
  const int c = tid * 4;
  const float4 gg = ((const float4*)g)[tid];
  const float4 bb = ((const float4*)bta)[tid];
  bf16* orow = out + (long)row * DM + c;
  orow[0] = (bf16)((v.x - mu) * rstd * gg.x + bb.x);
  orow[1] = (bf16)((v.y - mu) * rstd * gg.y + bb.y);
  orow[2] = (bf16)((v.z - mu) * rstd * gg.z + bb.z);
  orow[3] = (bf16)((v.w - mu) * rstd * gg.w + bb.w);
}

// ---------------- tiled WMMA GEMM: C[M,N] = A[M,K] x W[N,K]^T + bias ----------------
// Block: 256 threads = 8 waves, tile 128(M) x 128(N), BK=32, double-buffered LDS.
// Tile staging: TDM descriptors (wave 0) > async global->LDS > sync copies.
// Wave grid 2(M) x 4(N); each wave: 4x2 grid of 16x16 accumulators.
template<bool GELU, bool RES, bool WBF, bool WF>
__global__ __launch_bounds__(256)
void gemm_bf16_kernel(const bf16* __restrict__ A, const bf16* __restrict__ W,
                      const float* __restrict__ bias, const float* __restrict__ res,
                      float* __restrict__ outf, bf16* __restrict__ outb,
                      int N, int K) {
  __shared__ __align__(64) bf16 sA[2][128 * 32];
  __shared__ __align__(64) bf16 sB[2][128 * 32];

  const int tid  = threadIdx.x;
  const int lane = tid & 31;
  const int wid  = tid >> 5;
  const int m    = lane & 15;
  const int hlf  = lane >> 4;
  const int waveM = wid & 1;   // 0..1
  const int waveN = wid >> 1;  // 0..3
  const long tileM = (long)blockIdx.y * 128;
  const long tileN = (long)blockIdx.x * 128;

  v8f acc[4][2];
  #pragma unroll
  for (int i = 0; i < 4; ++i)
    #pragma unroll
    for (int j = 0; j < 2; ++j) acc[i][j] = vzero8();

  const int rL = tid >> 1;          // 0..127
  const int cL = (tid & 1) * 16;    // 0 or 16
  const bf16* gA = A + (tileM + rL) * (long)K + cL;
  const bf16* gB = W + (tileN + rL) * (long)K + cL;

#if HAVE_TDM
  const bf16* tA = A + tileM * (long)K;     // tile base rows
  const bf16* tB = W + tileN * (long)K;
  unsigned ldsA[2], ldsB[2];
  ldsA[0] = (unsigned)(unsigned long long)(AS3 char*)&sA[0][0];
  ldsA[1] = (unsigned)(unsigned long long)(AS3 char*)&sA[1][0];
  ldsB[0] = (unsigned)(unsigned long long)(AS3 char*)&sB[0][0];
  ldsB[1] = (unsigned)(unsigned long long)(AS3 char*)&sB[1][0];
#endif

  // stage one 32-wide K-slice of both tiles into LDS buffer `buf`
  auto stage = [&](int buf, int k0) {
#if HAVE_TDM
    if (wid == 0) {                         // TDM is a per-wave DMA (EXEC ignored)
      tdm_load_tile(tA + k0, ldsA[buf], K);
      tdm_load_tile(tB + k0, ldsB[buf], K);
    }
#elif HAVE_ASYNC_LDS
    AS1 v4i* ga = (AS1 v4i*)(gA + k0);
    AS1 v4i* gb = (AS1 v4i*)(gB + k0);
    AS3 v4i* la = (AS3 v4i*)&sA[buf][rL * 32 + cL];
    AS3 v4i* lb = (AS3 v4i*)&sB[buf][rL * 32 + cL];
    // IOFFSET is added to BOTH the global and the LDS address (ISA 8.4.4),
    // so the second 16B chunk is the same pointers with offset 16.
    __builtin_amdgcn_global_load_async_to_lds_b128(ga, la, 0, 0);
    __builtin_amdgcn_global_load_async_to_lds_b128(ga, la, 16, 0);
    __builtin_amdgcn_global_load_async_to_lds_b128(gb, lb, 0, 0);
    __builtin_amdgcn_global_load_async_to_lds_b128(gb, lb, 16, 0);
#else
    *(v16bf*)&sA[buf][rL * 32 + cL] = *(const v16bf*)(gA + k0);
    *(v16bf*)&sB[buf][rL * 32 + cL] = *(const v16bf*)(gB + k0);
#endif
  };

  auto stage_wait = [&]() {
#if HAVE_TDM
    if (wid == 0) wait_tensor0();
#elif HAVE_ASYNC_LDS
    wait_async0();
#endif
  };

  stage(0, 0);
  stage_wait();
  __syncthreads();

  for (int k0 = 0; k0 < K; k0 += 32) {
    const int buf = (k0 >> 5) & 1;
    if (k0 + 32 < K) {
      stage(buf ^ 1, k0 + 32);                 // overlap copy of next tile
      if (k0 + 64 < K) {                       // prefetch tile after next
        __builtin_prefetch(gA + k0 + 64, 0, 3);
        __builtin_prefetch(gB + k0 + 64, 0, 3);
      }
    }

    v16bf af[4], bfr[2];
    #pragma unroll
    for (int fm = 0; fm < 4; ++fm) {
      const int row = waveM * 64 + fm * 16 + m;
      af[fm] = join16(&sA[buf][row * 32 + hlf * 16], &sA[buf][row * 32 + hlf * 16 + 8]);
    }
    #pragma unroll
    for (int fn = 0; fn < 2; ++fn) {
      const int col = waveN * 32 + fn * 16 + m;
      bfr[fn] = *(const v16bf*)&sB[buf][col * 32 + hlf * 16];
    }
    #pragma unroll
    for (int fm = 0; fm < 4; ++fm)
      #pragma unroll
      for (int fn = 0; fn < 2; ++fn)
        acc[fm][fn] = wmma_bf16(af[fm], bfr[fn], acc[fm][fn]);

    stage_wait();        // staging writes into buf^1 have landed
    __syncthreads();     // all waves done writing buf^1 / reading buf
  }

  // epilogue: C layout: lane l, vgpr r -> (row = r + 8*(l>>4), col = l&15)
  #pragma unroll
  for (int fm = 0; fm < 4; ++fm) {
    #pragma unroll
    for (int fn = 0; fn < 2; ++fn) {
      const long col = tileN + waveN * 32 + fn * 16 + m;
      const float bv = bias[col];
      #pragma unroll
      for (int r = 0; r < 8; ++r) {
        const long row = tileM + waveM * 64 + fm * 16 + r + 8 * hlf;
        float v = acc[fm][fn][r] + bv;
        if (GELU) v = 0.5f * v * (1.0f + erff(v * 0.70710678118f));
        if (RES)  v += res[row * (long)N + col];
        if (WF)   outf[row * (long)N + col] = v;
        if (WBF)  outb[row * (long)N + col] = (bf16)v;
      }
    }
  }
}

// ---------------- fused causal flash attention ----------------
// grid = (SEQ/16, NH, BATCH), block = 32 (one wave handles a 16-query tile).
__global__ __launch_bounds__(32)
void attention_kernel(const bf16* __restrict__ qkv, bf16* __restrict__ o) {
  __shared__ __align__(64) bf16 sP[16 * 32];
  __shared__ __align__(64) bf16 sV[64 * 32];   // [hd][key] transposed V block

  const int lane = threadIdx.x & 31;
  const int m    = lane & 15;
  const int hlf  = lane >> 4;
  const int qt = blockIdx.x, h = blockIdx.y, b = blockIdx.z;
  const int q0 = qt * 16;
  const long rs = 3 * DM;  // qkv row stride (elements)

  const bf16* qp = qkv + (long)b * SEQ * rs + h * HD;
  const bf16* kp = qp + DM;
  const bf16* vp = qp + 2 * DM;

  // Q fragments (hd 0..31 and 32..63), A-matrix layout
  const bf16* qrow = qp + (long)(q0 + m) * rs;
  const v16bf qf0 = join16(qrow + hlf * 16,      qrow + 8 + hlf * 16);
  const v16bf qf1 = join16(qrow + 32 + hlf * 16, qrow + 40 + hlf * 16);

  v8f oacc[4];
  #pragma unroll
  for (int i = 0; i < 4; ++i) oacc[i] = vzero8();
  float mst[8], lst[8];
  #pragma unroll
  for (int r = 0; r < 8; ++r) { mst[r] = -1e30f; lst[r] = 0.f; }

  const int kend = q0 + 16;
  for (int kb = 0; kb < kend; kb += 32) {
    __syncthreads();
    // stage V block transposed: sV[hd][key]
    const bf16* vrow = vp + (long)(kb + lane) * rs;
    #pragma unroll 8
    for (int d = 0; d < 64; ++d) sV[d * 32 + lane] = vrow[d];

    if (kb + 32 < kend) {   // prefetch next K/V rows
      __builtin_prefetch(kp + (long)(kb + 32 + lane) * rs, 0, 3);
      __builtin_prefetch(vp + (long)(kb + 32 + lane) * rs, 0, 3);
    }

    // scores for keys kb..kb+15 (sc0) and kb+16..kb+31 (sc1)
    v8f sc0 = vzero8(), sc1 = vzero8();
    {
      const bf16* kr = kp + (long)(kb + m) * rs;
      const v16bf b0 = *(const v16bf*)(kr + hlf * 16);
      const v16bf b1 = *(const v16bf*)(kr + 32 + hlf * 16);
      sc0 = wmma_bf16(qf0, b0, sc0);
      sc0 = wmma_bf16(qf1, b1, sc0);
    }
    {
      const bf16* kr = kp + (long)(kb + 16 + m) * rs;
      const v16bf b0 = *(const v16bf*)(kr + hlf * 16);
      const v16bf b1 = *(const v16bf*)(kr + 32 + hlf * 16);
      sc1 = wmma_bf16(qf0, b0, sc1);
      sc1 = wmma_bf16(qf1, b1, sc1);
    }

    // online softmax over this 16x32 score tile
    float p0[8], p1[8], alpha[8];
    #pragma unroll
    for (int r = 0; r < 8; ++r) {
      const int rowg = q0 + r + 8 * hlf;
      const int key0 = kb + m, key1 = kb + 16 + m;
      float s0 = (key0 <= rowg) ? sc0[r] * 0.125f : -1e30f;
      float s1 = (key1 <= rowg) ? sc1[r] * 0.125f : -1e30f;
      float mx = fmaxf(s0, s1);
      mx = fmaxf(mx, __shfl_xor(mx, 1));
      mx = fmaxf(mx, __shfl_xor(mx, 2));
      mx = fmaxf(mx, __shfl_xor(mx, 4));
      mx = fmaxf(mx, __shfl_xor(mx, 8));
      const float mnew = fmaxf(mst[r], mx);
      alpha[r] = expf(mst[r] - mnew);
      const float e0 = expf(s0 - mnew);
      const float e1 = expf(s1 - mnew);
      float sm = e0 + e1;
      sm += __shfl_xor(sm, 1);
      sm += __shfl_xor(sm, 2);
      sm += __shfl_xor(sm, 4);
      sm += __shfl_xor(sm, 8);
      lst[r] = lst[r] * alpha[r] + sm;
      mst[r] = mnew;
      p0[r] = e0; p1[r] = e1;
    }

    // transpose P from C-layout to A-layout via LDS (bf16)
    #pragma unroll
    for (int r = 0; r < 8; ++r) {
      const int row = r + 8 * hlf;
      sP[row * 32 + m]      = (bf16)p0[r];
      sP[row * 32 + 16 + m] = (bf16)p1[r];
    }
    __syncthreads();
    const v16bf pf = join16(&sP[m * 32 + hlf * 16], &sP[m * 32 + hlf * 16 + 8]);

    #pragma unroll
    for (int fn = 0; fn < 4; ++fn) {
      const v16bf vf = *(const v16bf*)&sV[(fn * 16 + m) * 32 + hlf * 16];
      #pragma unroll
      for (int r = 0; r < 8; ++r) oacc[fn][r] *= alpha[r];
      oacc[fn] = wmma_bf16(pf, vf, oacc[fn]);
    }
  }

  // epilogue: normalize and write O (bf16) as [b*SEQ+row, h*64+hd]
  #pragma unroll
  for (int fn = 0; fn < 4; ++fn) {
    #pragma unroll
    for (int r = 0; r < 8; ++r) {
      const long row = (long)b * SEQ + q0 + r + 8 * hlf;
      const int col = h * HD + fn * 16 + m;
      const float v = oacc[fn][r] / lst[r];
      o[row * DM + col] = (bf16)v;
    }
  }
}

// ---------------- host-side orchestration ----------------
extern "C" void kernel_launch(void* const* d_in, const int* in_sizes, int n_in,
                              void* d_out, int out_size, void* d_ws, size_t ws_size,
                              hipStream_t stream) {
  const float* x     = (const float*)d_in[0];
  const float* ln1_g = (const float*)d_in[1];
  const float* ln1_b = (const float*)d_in[2];
  const float* qkv_w = (const float*)d_in[3];
  const float* qkv_b = (const float*)d_in[4];
  const float* out_w = (const float*)d_in[5];
  const float* out_b = (const float*)d_in[6];
  const float* ln2_g = (const float*)d_in[7];
  const float* ln2_b = (const float*)d_in[8];
  const float* fc1_w = (const float*)d_in[9];
  const float* fc1_b = (const float*)d_in[10];
  const float* fc2_w = (const float*)d_in[11];
  const float* fc2_b = (const float*)d_in[12];
  (void)in_sizes; (void)n_in; (void)out_size; (void)ws_size;
  float* out = (float*)d_out;

  char* ws = (char*)d_ws;
  size_t off = 0;
  auto wsalloc = [&](size_t bytes) -> void* {
    void* p = ws + off;
    off += (bytes + 255) & ~(size_t)255;
    return p;
  };
  bf16*  h_bf   = (bf16*) wsalloc((size_t)MTOT * DM * 2);       // LN1 / LN2 output (reused)
  bf16*  qkv_bf = (bf16*) wsalloc((size_t)MTOT * 3 * DM * 2);   // qkv activations
  bf16*  o_bf   = (bf16*) wsalloc((size_t)MTOT * DM * 2);       // attention output
  float* x1     = (float*)wsalloc((size_t)MTOT * DM * 4);       // residual after attention
  bf16*  ffn_bf = (bf16*) wsalloc((size_t)MTOT * DFF * 2);      // fc1 output
  bf16*  wqkv   = (bf16*) wsalloc((size_t)3 * DM * DM * 2);
  bf16*  wout   = (bf16*) wsalloc((size_t)DM * DM * 2);
  bf16*  wfc1   = (bf16*) wsalloc((size_t)DFF * DM * 2);
  bf16*  wfc2   = (bf16*) wsalloc((size_t)DM * DFF * 2);

  // 1. one-time weight conversion fp32 -> bf16
  auto cvt = [&](const float* src, bf16* dst, int n) {
    cvt_f32_bf16_kernel<<<(n / 4 + 255) / 256, 256, 0, stream>>>(src, dst, n);
  };
  cvt(qkv_w, wqkv, 3 * DM * DM);
  cvt(out_w, wout, DM * DM);
  cvt(fc1_w, wfc1, DFF * DM);
  cvt(fc2_w, wfc2, DM * DFF);

  // 2. LN1
  layernorm_bf16_kernel<<<MTOT, 256, 0, stream>>>(x, ln1_g, ln1_b, h_bf);

  // 3. qkv = h @ qkv_w^T + b  -> bf16
  gemm_bf16_kernel<false, false, true, false>
      <<<dim3(3 * DM / 128, MTOT / 128), 256, 0, stream>>>(
          h_bf, wqkv, qkv_b, nullptr, nullptr, qkv_bf, 3 * DM, DM);

  // 4. fused causal attention -> o_bf
  attention_kernel<<<dim3(SEQ / 16, NH, BATCH), 32, 0, stream>>>(qkv_bf, o_bf);

  // 5. x1 = x + o @ out_w^T + b  (fp32)
  gemm_bf16_kernel<false, true, false, true>
      <<<dim3(DM / 128, MTOT / 128), 256, 0, stream>>>(
          o_bf, wout, out_b, x, x1, nullptr, DM, DM);

  // 6. LN2 (reuse h_bf)
  layernorm_bf16_kernel<<<MTOT, 256, 0, stream>>>(x1, ln2_g, ln2_b, h_bf);

  // 7. ffn = gelu(h2 @ fc1_w^T + b) -> bf16
  gemm_bf16_kernel<true, false, true, false>
      <<<dim3(DFF / 128, MTOT / 128), 256, 0, stream>>>(
          h_bf, wfc1, fc1_b, nullptr, nullptr, ffn_bf, DFF, DM);

  // 8. out = x1 + ffn @ fc2_w^T + b (fp32, K=4096)
  gemm_bf16_kernel<false, true, false, true>
      <<<dim3(DM / 128, MTOT / 128), 256, 0, stream>>>(
          ffn_bf, wfc2, fc2_b, x1, out, nullptr, DM, DFF);
}
